// DraftBotModel_12120397709991
// MI455X (gfx1250) — matrix operand order repaired and checked by reading, not compiled, not run on hardware
//
#include <hip/hip_runtime.h>

typedef __attribute__((ext_vector_type(16))) _Float16 v16h;
typedef __attribute__((ext_vector_type(8)))  _Float16 v8h;
typedef __attribute__((ext_vector_type(8)))  float    v8f;

#define C_DIM 300
#define A_DIM 32
#define XSTRIDE 600
#define WAVES_PER_BLOCK 4
#define ROWS_PER_WAVE 16
#define ROWS_PER_BLOCK (WAVES_PER_BLOCK * ROWS_PER_WAVE)
#define CT_TILES 19   /* ceil(300/16) column tiles for GEMM2 */
#define KT_TILES 10   /* ceil(300/32) k tiles for GEMM1 */

__global__ __launch_bounds__(128)
void draftbot_fused_kernel(const float* __restrict__ X,
                           const float* __restrict__ W,
                           float* __restrict__ out)
{
    // W staged in f16, two layouts, zero-padded so WMMA K/N tiles read zeros OOB.
    __shared__ _Float16 W16[304 * 32];   // [c][a] row-major, rows 300..303 = 0  (GEMM2 B frags)
    __shared__ _Float16 Wt16[32 * 320];  // [a][c], cols 300..319 = 0            (GEMM1 B frags)
    __shared__ _Float16 apS[WAVES_PER_BLOCK][16 * 32]; // per-wave ap re-striping buffer

    const int tid  = threadIdx.x;
    const int lane = tid & 31;
    const int wave = tid >> 5;
    const int half = lane >> 4;    // 0: lanes 0-15, 1: lanes 16-31
    const int l15  = lane & 15;

    // ---- stage W -> LDS (f16, both layouts) ----
    for (int i = tid; i < C_DIM * A_DIM; i += 128) {
        int c = i >> 5, a = i & 31;
        _Float16 h = (_Float16)W[i];
        W16[i] = h;
        Wt16[a * 320 + c] = h;
    }
    for (int i = tid; i < 4 * 32; i += 128) W16[300 * 32 + i] = (_Float16)0.f;
    for (int i = tid; i < 32 * 20; i += 128) {
        int a = i / 20, c = 300 + (i % 20);
        Wt16[a * 320 + c] = (_Float16)0.f;
    }
    __syncthreads();

    const int rowBase = blockIdx.x * ROWS_PER_BLOCK + wave * ROWS_PER_WAVE;
    const int aRow    = rowBase + l15;                       // this lane's A-matrix row
    const float* cardsRow = X + (size_t)aRow * XSTRIDE + C_DIM;

    // ---- GEMM1: ap(16x32) = cards(16x300) @ W(300x32) ----
    v8f acc0 = {};   // a-dims 0..15
    v8f acc1 = {};   // a-dims 16..31
#pragma unroll
    for (int kt = 0; kt < KT_TILES; ++kt) {
        // A frag: contiguous layout => element i holds K = i + 16*half
        v16h aF;
#pragma unroll
        for (int i = 0; i < 16; ++i) {
            int kc = kt * 32 + i + 16 * half;
            float v = (kc < C_DIM) ? cardsRow[kc] : 0.f;
            aF[i] = (_Float16)v;
        }
        // B frag: contiguous 16 K per lane-half from transposed W
        int kb = kt * 32 + 16 * half;
        v16h bF0 = *(const v16h*)&Wt16[l15 * 320 + kb];
        v16h bF1 = *(const v16h*)&Wt16[(16 + l15) * 320 + kb];
        acc0 = __builtin_amdgcn_wmma_f32_16x16x32_f16(false, aF, false, bF0, (short)0, acc0, false, false);
        acc1 = __builtin_amdgcn_wmma_f32_16x16x32_f16(false, aF, false, bF1, (short)0, acc1, false, false);
    }

    // ap = acc + 1.0 ; re-stripe D-layout -> A-layout via per-wave LDS (DS ops are in-order per wave)
    _Float16* apW = &apS[wave][0];
#pragma unroll
    for (int j = 0; j < 8; ++j) {
        int m = j + 8 * half;               // D layout: VGPR j holds rows j / j+8
        apW[m * 32 + l15]      = (_Float16)(acc0[j] + 1.0f);
        apW[m * 32 + 16 + l15] = (_Float16)(acc1[j] + 1.0f);
    }
    v8h lo = *(const v8h*)&apW[l15 * 32 + 16 * half];        // K 0..7 (+16 if upper half)
    v8h hi = *(const v8h*)&apW[l15 * 32 + 8 + 16 * half];   // K 8..15 (+16 if upper half)
    v16h aF2;
#pragma unroll
    for (int i = 0; i < 8; ++i) { aF2[i] = lo[i]; aF2[i + 8] = hi[i]; }

    // ---- GEMM2: score tiles = ap(16x32) @ W^T(32x300), 19 tiles of 16 cols ----
    v8f sc[CT_TILES];
#pragma unroll
    for (int ct = 0; ct < CT_TILES; ++ct) {
        v16h bF = *(const v16h*)&W16[(ct * 16 + l15) * 32 + 16 * half];
        v8f z = {};
        sc[ct] = __builtin_amdgcn_wmma_f32_16x16x32_f16(false, aF2, false, bF, (short)0, z, false, false);
    }

    // ---- apply options mask; per-row max ----
    float mx[8];
#pragma unroll
    for (int j = 0; j < 8; ++j) mx[j] = -__builtin_inff();
#pragma unroll
    for (int ct = 0; ct < CT_TILES; ++ct) {
        int col = ct * 16 + l15;
#pragma unroll
        for (int j = 0; j < 8; ++j) {
            int r = rowBase + j + 8 * half;
            float x = 0.f;
            if (col < C_DIM) x = X[(size_t)r * XSTRIDE + col] * sc[ct][j];
            sc[ct][j] = x;
            mx[j] = fmaxf(mx[j], x);
        }
    }
    // cross-lane max within each 16-lane half (row lives in one half)
#pragma unroll
    for (int j = 0; j < 8; ++j) {
#pragma unroll
        for (int off = 1; off < 16; off <<= 1)
            mx[j] = fmaxf(mx[j], __shfl_xor(mx[j], off, 32));
    }

    // ---- sum of s*exp(x - b*s), then lse = log(sum) ----
    float sm[8];
#pragma unroll
    for (int j = 0; j < 8; ++j) sm[j] = 0.f;
#pragma unroll
    for (int ct = 0; ct < CT_TILES; ++ct) {
#pragma unroll
        for (int j = 0; j < 8; ++j) {
            float x = sc[ct][j];
            float s = (x > 0.f) ? 1.f : ((x < 0.f) ? -1.f : 0.f);
            sm[j] += s * __expf(x - mx[j] * s);
        }
    }
#pragma unroll
    for (int j = 0; j < 8; ++j) {
#pragma unroll
        for (int off = 1; off < 16; off <<= 1)
            sm[j] += __shfl_xor(sm[j], off, 32);
        sm[j] = __logf(sm[j]);   // lse, broadcast across the half-wave
    }

    // ---- out = s * (x - b*s - lse), coalesced per half-wave ----
#pragma unroll
    for (int ct = 0; ct < CT_TILES; ++ct) {
        int col = ct * 16 + l15;
        if (col < C_DIM) {
#pragma unroll
            for (int j = 0; j < 8; ++j) {
                int r = rowBase + j + 8 * half;
                float x = sc[ct][j];
                float s = (x > 0.f) ? 1.f : ((x < 0.f) ? -1.f : 0.f);
                out[(size_t)r * C_DIM + col] = s * (x - mx[j] * s - sm[j]);
            }
        }
    }
}

extern "C" void kernel_launch(void* const* d_in, const int* in_sizes, int n_in,
                              void* d_out, int out_size, void* d_ws, size_t ws_size,
                              hipStream_t stream) {
    const float* X = (const float*)d_in[0];   // (B, 600) f32
    const float* W = (const float*)d_in[1];   // (300, 32) f32
    float* out = (float*)d_out;               // (B, 300) f32

    int Brows = in_sizes[0] / XSTRIDE;        // 65536
    dim3 grid(Brows / ROWS_PER_BLOCK);        // 1024 blocks
    dim3 block(128);                          // 4 waves (wave32)
    hipLaunchKernelGGL(draftbot_fused_kernel, grid, block, 0, stream, X, W, out);
}
